// MambaLayer_2508260901096
// MI455X (gfx1250) — matrix operand order repaired and checked by reading, not compiled, not run on hardware
//
#include <hip/hip_runtime.h>

// ---------------------------------------------------------------------------
// Mamba layer forward for gfx1250 (MI455X): bf16 WMMA GEMMs + wave32 scan.
// ---------------------------------------------------------------------------

typedef __bf16 bf16;
typedef __attribute__((ext_vector_type(8)))  bf16  v8bf;
typedef __attribute__((ext_vector_type(16))) bf16  v16bf;
typedef __attribute__((ext_vector_type(8)))  float v8f;

#define D_MODEL 1024
#define D_INNER 2048
#define D_STATE 16
#define DT_RANK 64
#define SEQLEN  2048
#define NROWS   4096            // B * L
#define XZ_N    4096            // 2 * D_INNER

// ---------------- workspace layout (bytes, all 256-aligned) ----------------
static const size_t WS_HS_BF    = 0;                    // 4096*1024 bf16
static const size_t WS_WIN      = 8388608;              // in_proj^T  [4096,1024] bf16
static const size_t WS_XZ       = 16777216;             // xz f32 [4096,4096]
static const size_t WS_XCONV_F  = 83886080;             // conv+silu f32 [4096,2048]
static const size_t WS_XCONV_BF = 117440512;            // same in bf16
static const size_t WS_WXP      = 134217728;            // x_proj^T [96,2048] bf16
static const size_t WS_XDBL     = 134610944;            // x_dbl f32 [4096,96]
static const size_t WS_DTIN     = 136183808;            // dt input bf16 [4096,64]
static const size_t WS_WDT      = 136708096;            // dt_proj^T [2048,64] bf16
static const size_t WS_DT       = 136970240;            // dt f32 [4096,2048] (in-place softplus)
static const size_t WS_YBF      = 170524672;            // gated y bf16 [4096,2048]
static const size_t WS_WOUT     = 187301888;            // out_proj^T [1024,2048] bf16
static const size_t WS_A        = 191496192;            // A = -exp(A_log) f32 [2048,16]

// ---------------------------------------------------------------------------
// WMMA helper: D = A(16x32 bf16) * B(32x16 bf16) + C(f32)
// ---------------------------------------------------------------------------
__device__ __forceinline__ v8f wmma_bf16(v16bf a, v16bf b, v8f c) {
  return __builtin_amdgcn_wmma_f32_16x16x32_bf16(
      /*neg_a=*/false, a, /*neg_b=*/false, b,
      /*c_mod=*/(short)0, c, /*reuse_a=*/false, /*reuse_b=*/false);
}

__device__ __forceinline__ v16bf cat16(v8bf lo, v8bf hi) {
  return __builtin_shufflevector(lo, hi, 0,1,2,3,4,5,6,7,8,9,10,11,12,13,14,15);
}

// One wave-step worth of operand fragments (A 32xK-tile, B Kx64-tile).
struct Frags {
  v16bf a0, a1;
  v16bf b[4];
};

__device__ __forceinline__ void load_frags(
    Frags& f, const bf16* __restrict__ arow0, const bf16* __restrict__ arow1,
    const bf16* const* brow, int kb)
{
  f.a0 = cat16(*(const v8bf*)(arow0 + kb), *(const v8bf*)(arow0 + kb + 16));
  f.a1 = cat16(*(const v8bf*)(arow1 + kb), *(const v8bf*)(arow1 + kb + 16));
  #pragma unroll
  for (int t = 0; t < 4; ++t)
    f.b[t] = cat16(*(const v8bf*)(brow[t] + kb), *(const v8bf*)(brow[t] + kb + 16));
}

__device__ __forceinline__ void mma_frags(v8f acc[2][4], const Frags& f)
{
  #pragma unroll
  for (int t = 0; t < 4; ++t) {
    acc[0][t] = wmma_bf16(f.a0, f.b[t], acc[0][t]);
    acc[1][t] = wmma_bf16(f.a1, f.b[t], acc[1][t]);
  }
}

// ---------------------------------------------------------------------------
// Generic bf16 GEMM: C[M,N] f32 = A[M,K] bf16 (row major) x BT[N,K] bf16.
// Block = 256 threads = 8 waves (2 M-waves x 4 N-waves); wave tile 32x64
// (2 M-tiles x 4 N-tiles of 16x16).  Block tile 64x256.
// grid = (M/64, ceil(N/256)).  M%64==0, K%64==0, N%16==0 required.
// Out-of-range N tiles clamp their B pointer to tile 0 (valid memory) and
// skip the store.  K loop is unrolled x2 with two named fragment sets in
// ping-pong (no rotation copies); loads for one set issue before the other
// set's 8 WMMAs, hiding VMEM latency behind the WMMA issue stream.
// ---------------------------------------------------------------------------
__global__ __launch_bounds__(256) void gemm_bf16_kernel(
    const bf16* __restrict__ A, int lda,
    const bf16* __restrict__ BT, int ldb,
    float* __restrict__ C, int ldc,
    int N, int K)
{
  const int tid  = threadIdx.x;
  const int lane = tid & 31;
  const int wid  = tid >> 5;
  const int m0   = blockIdx.x * 64  + (wid >> 2) * 32;
  const int n0   = blockIdx.y * 256 + (wid & 3)  * 64;
  const int mr   = lane & 15;      // A row / B column within 16-tile
  const int half = lane >> 4;      // K-half selector

  v8f acc[2][4];
  #pragma unroll
  for (int i = 0; i < 2; ++i)
    #pragma unroll
    for (int t = 0; t < 4; ++t)
      acc[i][t] = {};

  // A-fragment layout: lane m=lane%16; K = half*8+{0..7} (v0-3), 16+half*8+{0..7} (v4-7)
  const bf16* arow0 = A + (size_t)(m0 + mr) * lda + half * 8;
  const bf16* arow1 = arow0 + (size_t)16 * lda;

  // B-fragment layout: lane n=lane%16; K = half*8+{0..7} (v0-3), 16+half*8+{0..7} (v4-7) (from W^T)
  const bf16* brow[4];
  bool nok[4];
  #pragma unroll
  for (int t = 0; t < 4; ++t) {
    const int nt = n0 + t * 16;
    nok[t] = (nt < N);
    brow[t] = BT + (size_t)((nok[t] ? nt : 0) + mr) * ldb + half * 8;
  }

  // ---- software pipeline: two fragment sets in ping-pong (K % 64 == 0) ----
  Frags f0, f1;
  load_frags(f0, arow0, arow1, brow, 0);

  for (int kb = 0; kb < K; kb += 64) {
    // prefetch the streaming A operand ahead (weights sit in the 192MB L2);
    // speculative prefetch past the end is silently dropped
    __builtin_prefetch(arow0 + kb + 256, 0, 1);
    __builtin_prefetch(arow1 + kb + 256, 0, 1);

    load_frags(f1, arow0, arow1, brow, kb + 32);
    mma_frags(acc, f0);

    const int k2 = (kb + 64 < K) ? kb + 64 : 0;   // tail: harmless reload of step 0
    load_frags(f0, arow0, arow1, brow, k2);
    mma_frags(acc, f1);
  }

  // C/D layout: lane n=lane%16, VGPR r -> row m = r + 8*half
  const int cn = lane & 15;
  #pragma unroll
  for (int t = 0; t < 4; ++t) {
    if (!nok[t]) continue;
    #pragma unroll
    for (int i = 0; i < 2; ++i) {
      float* crow = C + (size_t)(m0 + i * 16 + half * 8) * ldc + n0 + t * 16 + cn;
      #pragma unroll
      for (int r = 0; r < 8; ++r)
        crow[(size_t)r * ldc] = acc[i][t][r];
    }
  }
}

// ---------------------------------------------------------------------------
// Weight prep: f32 [R,C] -> bf16 transposed [C,R]
// ---------------------------------------------------------------------------
__global__ __launch_bounds__(256) void conv_transpose_bf16_kernel(
    const float* __restrict__ in, bf16* __restrict__ out, int R, int Cc)
{
  int i = blockIdx.x * 256 + threadIdx.x;
  if (i >= R * Cc) return;
  int r = i / Cc, c = i % Cc;
  out[(size_t)c * R + r] = (bf16)in[i];
}

__global__ __launch_bounds__(256) void negexp_kernel(
    const float* __restrict__ in, float* __restrict__ out, int n)
{
  int i = blockIdx.x * 256 + threadIdx.x;
  if (i < n) out[i] = -__expf(in[i]);
}

// ---------------------------------------------------------------------------
// Fused add + RMSNorm: residual_out = hs+res (f32, to d_out), hs_bf = norm (bf16)
// One block (256 thr) per row of D_MODEL=1024.
// ---------------------------------------------------------------------------
__global__ __launch_bounds__(256) void addnorm_kernel(
    const float* __restrict__ hs, const float* __restrict__ res,
    const float* __restrict__ w, float* __restrict__ res_out,
    bf16* __restrict__ hs_bf)
{
  __shared__ float sm[256];
  const int tid = threadIdx.x;
  const size_t base = (size_t)blockIdx.x * D_MODEL + tid * 4;
  float v[4];
  float ss = 0.f;
  #pragma unroll
  for (int j = 0; j < 4; ++j) {
    float t = hs[base + j] + res[base + j];
    v[j] = t;
    ss += t * t;
    res_out[base + j] = t;
  }
  sm[tid] = ss;
  __syncthreads();
  for (int s = 128; s > 0; s >>= 1) {
    if (tid < s) sm[tid] += sm[tid + s];
    __syncthreads();
  }
  const float scale = rsqrtf(sm[0] * (1.0f / D_MODEL) + 1e-5f);
  #pragma unroll
  for (int j = 0; j < 4; ++j)
    hs_bf[base + j] = (bf16)(v[j] * scale * w[tid * 4 + j]);
}

// ---------------------------------------------------------------------------
// Causal depthwise conv1d (K=4) + bias + SiLU.  x = xz[:, :D_INNER].
// ---------------------------------------------------------------------------
__global__ __launch_bounds__(256) void conv_silu_kernel(
    const float* __restrict__ xz, const float* __restrict__ cw,
    const float* __restrict__ cb, float* __restrict__ xf, bf16* __restrict__ xb)
{
  int i = blockIdx.x * 256 + threadIdx.x;      // over NROWS * D_INNER
  int d   = i & (D_INNER - 1);
  int row = i >> 11;                            // b*L + l
  int l   = row & (SEQLEN - 1);
  int bL  = row - l;
  float acc = cb[d];
  #pragma unroll
  for (int k = 0; k < 4; ++k) {
    int ls = l + k - 3;
    if (ls >= 0) acc += xz[((size_t)(bL + ls) << 12) + d] * cw[d * 4 + k];
  }
  float s = acc / (1.0f + __expf(-acc));
  xf[i] = s;
  xb[i] = (bf16)s;
}

// Extract dt-rank slice of x_dbl to bf16 GEMM input.
__global__ __launch_bounds__(256) void dtin_kernel(
    const float* __restrict__ xdbl, bf16* __restrict__ out)
{
  int i = blockIdx.x * 256 + threadIdx.x;      // over NROWS * DT_RANK
  int row = i >> 6, j = i & 63;
  out[i] = (bf16)xdbl[(size_t)row * 96 + j];
}

// In-place softplus(dt + bias)
__global__ __launch_bounds__(256) void softplus_kernel(
    float* __restrict__ dt, const float* __restrict__ b)
{
  int i = blockIdx.x * 256 + threadIdx.x;      // over NROWS * D_INNER
  float v = dt[i] + b[i & (D_INNER - 1)];
  dt[i] = (v > 20.0f) ? v : log1pf(__expf(v));
}

// ---------------------------------------------------------------------------
// Selective scan.  grid = B * D_INNER/16 blocks, 256 threads.
// Each wave: 2 channels; lane = d_local*16 + n (n = state index 0..15).
// Recurrence h = h*exp(dt*A) + dt*x*B ; y = sum_n h*C ; then D-skip + z-gate.
// ---------------------------------------------------------------------------
__global__ __launch_bounds__(256) void scan_kernel(
    const float* __restrict__ dt, const float* __restrict__ xc,
    const float* __restrict__ xdbl, const float* __restrict__ xz,
    const float* __restrict__ Abuf, const float* __restrict__ Dskip,
    bf16* __restrict__ ybf)
{
  const int tid  = threadIdx.x;
  const int lane = tid & 31;
  const int n    = lane & 15;
  const int b    = blockIdx.x >> 7;                    // 128 channel-blocks / batch
  const int cb   = blockIdx.x & 127;
  const int d    = cb * 16 + (tid >> 5) * 2 + (lane >> 4);
  const float Ad = Abuf[d * D_STATE + n];
  const float Dd = Dskip[d];
  float h = 0.f;
  size_t row = (size_t)b * SEQLEN;
  for (int l = 0; l < SEQLEN; ++l, ++row) {
    const size_t r2k = row << 11;
    float dtv = dt[r2k + d];
    float xv  = xc[r2k + d];
    float Bv  = xdbl[row * 96 + DT_RANK + n];
    float Cv  = xdbl[row * 96 + DT_RANK + D_STATE + n];
    h = h * __expf(dtv * Ad) + dtv * xv * Bv;
    float y = h * Cv;
    #pragma unroll
    for (int off = 8; off >= 1; off >>= 1) y += __shfl_xor(y, off, 16);
    if (n == 0) {
      float zv = xz[(row << 12) + D_INNER + d];
      float g  = zv / (1.0f + __expf(-zv));
      ybf[r2k + d] = (bf16)((y + xv * Dd) * g);
    }
  }
}

// ---------------------------------------------------------------------------
extern "C" void kernel_launch(void* const* d_in, const int* in_sizes, int n_in,
                              void* d_out, int out_size, void* d_ws, size_t ws_size,
                              hipStream_t stream) {
  (void)in_sizes; (void)n_in; (void)out_size; (void)ws_size;

  const float* hs        = (const float*)d_in[0];
  const float* res       = (const float*)d_in[1];
  const float* norm_w    = (const float*)d_in[2];
  const float* in_proj_w = (const float*)d_in[3];   // [1024,4096]
  const float* conv_w    = (const float*)d_in[4];   // [2048,4]
  const float* conv_b    = (const float*)d_in[5];
  const float* x_proj_w  = (const float*)d_in[6];   // [2048,96]
  const float* dt_proj_w = (const float*)d_in[7];   // [64,2048]
  const float* dt_proj_b = (const float*)d_in[8];
  const float* A_log     = (const float*)d_in[9];   // [2048,16]
  const float* D_skip    = (const float*)d_in[10];
  const float* out_proj_w= (const float*)d_in[11];  // [2048,1024]

  float* out     = (float*)d_out;                         // [4096,1024]
  float* res_out = out + (size_t)NROWS * D_MODEL;         // [4096,1024]

  char* ws = (char*)d_ws;
  bf16*  HS   = (bf16*) (ws + WS_HS_BF);
  bf16*  WIN  = (bf16*) (ws + WS_WIN);
  float* XZ   = (float*)(ws + WS_XZ);
  float* XCF  = (float*)(ws + WS_XCONV_F);
  bf16*  XCB  = (bf16*) (ws + WS_XCONV_BF);
  bf16*  WXP  = (bf16*) (ws + WS_WXP);
  float* XDBL = (float*)(ws + WS_XDBL);
  bf16*  DTIN = (bf16*) (ws + WS_DTIN);
  bf16*  WDT  = (bf16*) (ws + WS_WDT);
  float* DT   = (float*)(ws + WS_DT);
  bf16*  YBF  = (bf16*) (ws + WS_YBF);
  bf16*  WOUT = (bf16*) (ws + WS_WOUT);
  float* ABUF = (float*)(ws + WS_A);

  // --- weight prep: bf16 transposed copies + A = -exp(A_log) ---
  conv_transpose_bf16_kernel<<<(1024*4096+255)/256, 256, 0, stream>>>(in_proj_w,  WIN,  1024, 4096);
  conv_transpose_bf16_kernel<<<(2048*96  +255)/256, 256, 0, stream>>>(x_proj_w,   WXP,  2048, 96);
  conv_transpose_bf16_kernel<<<(64*2048  +255)/256, 256, 0, stream>>>(dt_proj_w,  WDT,  64,   2048);
  conv_transpose_bf16_kernel<<<(2048*1024+255)/256, 256, 0, stream>>>(out_proj_w, WOUT, 2048, 1024);
  negexp_kernel<<<(2048*16+255)/256, 256, 0, stream>>>(A_log, ABUF, 2048*16);

  // --- fused add + RMSNorm ---
  addnorm_kernel<<<NROWS, 256, 0, stream>>>(hs, res, norm_w, res_out, HS);

  // --- in_proj: [4096,1024] x [1024,4096] -> xz ---
  gemm_bf16_kernel<<<dim3(NROWS/64, XZ_N/256), 256, 0, stream>>>(
      HS, D_MODEL, WIN, D_MODEL, XZ, XZ_N, XZ_N, D_MODEL);

  // --- causal conv + SiLU ---
  conv_silu_kernel<<<(NROWS*D_INNER)/256, 256, 0, stream>>>(XZ, conv_w, conv_b, XCF, XCB);

  // --- x_proj: [4096,2048] x [2048,96] -> x_dbl ---
  gemm_bf16_kernel<<<dim3(NROWS/64, 1), 256, 0, stream>>>(
      XCB, D_INNER, WXP, D_INNER, XDBL, 96, 96, D_INNER);

  // --- dt_proj: [4096,64] x [64,2048] -> dt, then softplus(+bias) in place ---
  dtin_kernel<<<(NROWS*DT_RANK)/256, 256, 0, stream>>>(XDBL, DTIN);
  gemm_bf16_kernel<<<dim3(NROWS/64, D_INNER/256), 256, 0, stream>>>(
      DTIN, DT_RANK, WDT, DT_RANK, DT, D_INNER, D_INNER, DT_RANK);
  softplus_kernel<<<(NROWS*D_INNER)/256, 256, 0, stream>>>(DT, dt_proj_b);

  // --- selective scan + D-skip + z-gate -> bf16 y ---
  scan_kernel<<<2 * (D_INNER/16), 256, 0, stream>>>(DT, XCF, XDBL, XZ, ABUF, D_skip, YBF);

  // --- out_proj: [4096,2048] x [2048,1024] -> out ---
  gemm_bf16_kernel<<<dim3(NROWS/64, D_MODEL/256), 256, 0, stream>>>(
      YBF, D_INNER, WOUT, D_INNER, out, D_MODEL, D_MODEL, D_INNER);
}
